// Decoder_66039417143669
// MI455X (gfx1250) — compile-verified, matrix operand order checked by reading, not run on hardware
//
#include <hip/hip_runtime.h>

// Problem constants (from reference): B=1024, S=256, H=128, O=7, T=512
#define Bn 1024
#define Sn 256
#define Hn 128
#define On 7
#define Tn 512
#define HP 136          // padded LDS row stride in halves (272B -> conflict-free 16B row loads)

typedef __attribute__((ext_vector_type(16))) _Float16 v16h;
typedef __attribute__((ext_vector_type(8)))  _Float16 v8h;
typedef __attribute__((ext_vector_type(8)))  float    v8f;

__device__ __forceinline__ float sigmoidf_(float x) { return 1.0f / (1.0f + __expf(-x)); }
__device__ __forceinline__ float tanhf_(float x)    { return 1.0f - 2.0f / (1.0f + __expf(2.0f * x)); }

// Assemble a 16x32 fp16 WMMA A-fragment for this lane from a row-major LDS h plane.
// Lane L (<16): row=L, elems 0..7 = K[k0..k0+7], elems 8..15 = K[k0+16..k0+23]; k0 already
// includes the +8 bias for lanes 16..31.
__device__ __forceinline__ v16h load_a(const _Float16* plane, int row, int k0) {
  v8h a = *(const v8h*)(plane + row * HP + k0);
  v8h b = *(const v8h*)(plane + row * HP + k0 + 16);
  v16h r;
#pragma unroll
  for (int i = 0; i < 8; ++i) { r[i] = a[i]; r[i + 8] = b[i]; }
  return r;
}

// ---------------------------------------------------------------------------
// Prep: fold Wc = W_ih + W_hh, split into fp16 hi/lo planes, pre-swizzle into
// WMMA B-fragment order so the main kernel's register loads are fully coalesced.
// Fragment f layout in d_ws (as halves): [(fragId*32 + lane)*16 + e]
//   fragId 0..127   : hi plane of Wc   (nt*4 + kf)
//   fragId 128..255 : lo plane of Wc
//   fragId 256..259 : hi plane of W_out (padded 7->16 cols), kf
//   fragId 260..263 : lo plane of W_out
// ---------------------------------------------------------------------------
__global__ void prep_kernel(const float* __restrict__ W_ih, const float* __restrict__ W_hh,
                            const float* __restrict__ W_out, _Float16* __restrict__ frags) {
  int t = blockIdx.x * blockDim.x + threadIdx.x;
  int e    = t & 15;
  int lane = (t >> 4) & 31;
  int f    = t >> 9;
  if (f >= 132) return;

  int kf;
  float w;
  if (f < 128) {
    int nt = f >> 2;
    kf = f & 3;
    int N = nt * 16 + (lane & 15);
    int kb = kf * 32 + ((lane < 16) ? 0 : 8);
    int K = kb + e + ((e < 8) ? 0 : 8);
    w = W_ih[N * Hn + K] + W_hh[N * Hn + K];
  } else {
    kf = f - 128;
    int N = lane & 15;
    int kb = kf * 32 + ((lane < 16) ? 0 : 8);
    int K = kb + e + ((e < 8) ? 0 : 8);
    w = (N < On) ? W_out[N * Hn + K] : 0.0f;
  }
  _Float16 hi = (_Float16)w;
  _Float16 lo = (_Float16)(w - (float)hi);
  int fh = (f < 128) ? f : (256 + kf);
  int fl = (f < 128) ? (128 + f) : (260 + kf);
  frags[(fh * 32 + lane) * 16 + e] = hi;
  frags[(fl * 32 + lane) * 16 + e] = lo;
}

// ---------------------------------------------------------------------------
// Main persistent LSTM rollout. One WG = 16 batch rows, 8 waves.
// Wave w owns hidden cols [w*16, w*16+16) and computes all 4 gates for them,
// so c-update / activations are register-only. h is broadcast through a
// double-buffered LDS tile; one barrier per step. Gate chains start with
// WMMA SRC2 = inline 0; biases are folded into the activation stage.
// ---------------------------------------------------------------------------
__global__ __launch_bounds__(256, 1) void lstm_kernel(
    const float* __restrict__ ctx, const float* __restrict__ b_ih,
    const float* __restrict__ b_hh, const float* __restrict__ b_out,
    const _Float16* __restrict__ frags, float* __restrict__ out) {
  __shared__ _Float16 hbuf[2][2][16 * HP];  // [buffer][hi/lo plane][row*HP + col]

  const int tid = threadIdx.x;
  const int lane = tid & 31;
  const int wv = tid >> 5;
  const int wg = blockIdx.x;
  const int ln16 = lane & 15;
  const int hihalf = lane >> 4;       // 0 or 1
  const int kbo = hihalf ? 8 : 0;     // K-base bias for upper half-wave (A & B layouts)

  // ---- h0 = context_seq[:, S-1, :], split hi/lo into LDS buffer 0 ----
  {
    int r = tid >> 4;                 // 0..15  (batch row within tile)
    int k0 = (tid & 15) * 8;          // 0..120
    const float* src = ctx + ((size_t)(wg * 16 + r) * Sn + (Sn - 1)) * Hn + k0;
#pragma unroll
    for (int i = 0; i < 8; ++i) {
      float v = src[i];
      _Float16 h = (_Float16)v;
      _Float16 l = (_Float16)(v - (float)h);
      hbuf[0][0][r * HP + k0 + i] = h;
      hbuf[0][1][r * HP + k0 + i] = l;
    }
  }

  // ---- persistent weight fragments in VGPRs ----
  const v16h* fr = (const v16h*)frags;
  v16h bwh[4][4], bwl[4][4];          // [gate][kfrag]
#pragma unroll
  for (int g = 0; g < 4; ++g) {
    int nt = g * 8 + wv;
#pragma unroll
    for (int kf = 0; kf < 4; ++kf) {
      bwh[g][kf] = fr[(nt * 4 + kf) * 32 + lane];
      bwl[g][kf] = fr[(128 + nt * 4 + kf) * 32 + lane];
    }
  }
  v16h owh[4], owl[4];                // W_out fragments (wave 0 only)
  if (wv == 0) {
#pragma unroll
    for (int kf = 0; kf < 4; ++kf) {
      owh[kf] = fr[(256 + kf) * 32 + lane];
      owl[kf] = fr[(260 + kf) * 32 + lane];
    }
  }

  // ---- biases (added post-matmul in the activation stage) ----
  float bias[4];
#pragma unroll
  for (int g = 0; g < 4; ++g) {
    int col = (g * 8 + wv) * 16 + ln16;
    bias[g] = b_ih[col] + b_hh[col];
  }
  float ob = 0.0f;
  if (wv == 0 && ln16 < On) ob = b_out[ln16];

  // ---- cell state: (row r + 8*hihalf, hidden col wv*16+ln16) per lane-slot ----
  float c[8];
#pragma unroll
  for (int r = 0; r < 8; ++r) c[r] = 0.0f;

  __syncthreads();

  const int colw = wv * 16 + ln16;
  const size_t outbase = (size_t)wg * 16 * Tn * On;

  for (int t = 0; t < Tn; ++t) {
    const _Float16* hc0 = hbuf[t & 1][0];
    const _Float16* hc1 = hbuf[t & 1][1];
    _Float16* hn0 = hbuf[(t + 1) & 1][0];
    _Float16* hn1 = hbuf[(t + 1) & 1][1];

    v8f acc[4];

    // gates = (h_hi + h_lo) @ (W_hi + W_lo)^T  ~= hi*Whi + hi*Wlo + lo*Whi
    // First product of each chain uses SRC2 = inline 0 (no acc-init movs).
#pragma unroll
    for (int kf = 0; kf < 4; ++kf) {
      int k0 = kf * 32 + kbo;
      v16h ah = load_a(hc0, ln16, k0);
      v16h al = load_a(hc1, ln16, k0);
#pragma unroll
      for (int g = 0; g < 4; ++g) {
        v8f a0;
        if (kf == 0) {
          v8f z = {};
          a0 = __builtin_amdgcn_wmma_f32_16x16x32_f16(false, ah, false, bwh[g][kf],
                                                      (short)0, z, false, false);
        } else {
          a0 = __builtin_amdgcn_wmma_f32_16x16x32_f16(false, ah, false, bwh[g][kf],
                                                      (short)0, acc[g], false, false);
        }
        a0 = __builtin_amdgcn_wmma_f32_16x16x32_f16(false, ah, false, bwl[g][kf],
                                                    (short)0, a0, false, false);
        acc[g] = __builtin_amdgcn_wmma_f32_16x16x32_f16(false, al, false, bwh[g][kf],
                                                        (short)0, a0, false, false);
      }
    }

    // pointwise LSTM cell (bias folded in here), register-resident c;
    // write split h_new to the next buffer
#pragma unroll
    for (int r = 0; r < 8; ++r) {
      float gi = acc[0][r] + bias[0];
      float gf = acc[1][r] + bias[1];
      float gg = acc[2][r] + bias[2];
      float go = acc[3][r] + bias[3];
      float cn = sigmoidf_(gf) * c[r] + sigmoidf_(gi) * tanhf_(gg);
      float hn = sigmoidf_(go) * tanhf_(cn);
      c[r] = cn;
      _Float16 hh = (_Float16)hn;
      _Float16 hl = (_Float16)(hn - (float)hh);
      int row = r + 8 * hihalf;
      hn0[row * HP + colw] = hh;
      hn1[row * HP + colw] = hl;
    }

    __syncthreads();

    // wave 0: pred_t = h_new @ W_out^T + b_out via one padded WMMA tile
    if (wv == 0) {
      v8f oacc;
#pragma unroll
      for (int kf = 0; kf < 4; ++kf) {
        int k0 = kf * 32 + kbo;
        v16h ah = load_a(hn0, ln16, k0);
        v16h al = load_a(hn1, ln16, k0);
        v8f a0;
        if (kf == 0) {
          v8f z = {};
          a0 = __builtin_amdgcn_wmma_f32_16x16x32_f16(false, ah, false, owh[kf],
                                                      (short)0, z, false, false);
        } else {
          a0 = __builtin_amdgcn_wmma_f32_16x16x32_f16(false, ah, false, owh[kf],
                                                      (short)0, oacc, false, false);
        }
        a0 = __builtin_amdgcn_wmma_f32_16x16x32_f16(false, ah, false, owl[kf],
                                                    (short)0, a0, false, false);
        oacc = __builtin_amdgcn_wmma_f32_16x16x32_f16(false, al, false, owh[kf],
                                                      (short)0, a0, false, false);
      }
      if (ln16 < On) {
#pragma unroll
        for (int r = 0; r < 8; ++r) {
          int row = r + 8 * hihalf;
          out[outbase + (size_t)row * Tn * On + (size_t)t * On + ln16] = oacc[r] + ob;
        }
      }
    }
  }
}

extern "C" void kernel_launch(void* const* d_in, const int* in_sizes, int n_in,
                              void* d_out, int out_size, void* d_ws, size_t ws_size,
                              hipStream_t stream) {
  const float* ctx   = (const float*)d_in[0];
  const float* W_ih  = (const float*)d_in[1];
  const float* W_hh  = (const float*)d_in[2];
  const float* b_ih  = (const float*)d_in[3];
  const float* b_hh  = (const float*)d_in[4];
  const float* W_out = (const float*)d_in[5];
  const float* b_out = (const float*)d_in[6];
  _Float16* frags = (_Float16*)d_ws;   // 264 frags * 512 halves * 2B = ~270 KB

  prep_kernel<<<264, 256, 0, stream>>>(W_ih, W_hh, W_out, frags);
  lstm_kernel<<<Bn / 16, 256, 0, stream>>>(ctx, b_ih, b_hh, b_out, frags, (float*)d_out);
}
